// UltralyticsDetect_65773129171251
// MI455X (gfx1250) — matrix operand order repaired and checked by reading, not compiled, not run on hardware
//
#include <hip/hip_runtime.h>

typedef __attribute__((ext_vector_type(16))) _Float16 v16h;
typedef __attribute__((ext_vector_type(8)))  float    v8f;
typedef __attribute__((ext_vector_type(4)))  float    v4f;

#define A_TOTAL 8400
#define WPB     8          // waves per block

// Packed-weight layout (per level), in _Float16 units:
//   group g = ((m * nkc + kc) * 32 + lane), each group = 32 halves: [16 x hi][16 x lo]
// Level sizes (halves): L0 (nkc=8): 73728, L1/L2 (nkc=16): 147456 each. Total 737280 bytes.
#define WP0_OFF 0
#define WP1_OFF 73728
#define WP2_OFF 221184

// ---------------------------------------------------------------------------
// Pre-pass: split one level's weights into f16 hi/lo, stored in A-fragment
// lane order so the GEMM loop does pure b128 loads with zero conversion VALU.
// ---------------------------------------------------------------------------
__global__ __launch_bounds__(256)
void pack_weights(const float* __restrict__ wb, const float* __restrict__ wc,
                  _Float16* __restrict__ dst, int C)
{
    const int nkc   = C / 32;
    const int total = 9 * nkc * 32;          // groups
    int g = blockIdx.x * blockDim.x + threadIdx.x;
    if (g >= total) return;

    const int lane = g & 31;
    const int kc   = (g >> 5) % nkc;
    const int m    = (g >> 5) / nkc;
    const int col  = lane & 15;
    const int hiK  = lane >> 4;

    const int o = m * 16 + col;
    const float* wrow = (m < 4) ? (wb + (size_t)o * C)
                                : (wc + (size_t)(o - 64) * C);
    const int kbase = kc * 32 + hiK * 8;

    _Float16* d = dst + (size_t)g * 32;
    #pragma unroll
    for (int e = 0; e < 16; ++e) {
        int k = kbase + ((e < 8) ? e : (e + 8));   // A-fragment K mapping
        float v = wrow[k];
        _Float16 h = (_Float16)v;
        d[e]      = h;
        d[16 + e] = (_Float16)(v - (float)h);
    }
}

// ---------------------------------------------------------------------------
// Main detect head, specialized per FPN level so every stride / offset is a
// compile-time immediate: WMMA GEMM (f16 hi/lo split, f32 acc) + DFL + decode.
// ---------------------------------------------------------------------------
template<int C, int HW, int W, int STRIDE_I, int LSTART, int NTILES>
__global__ __launch_bounds__(256)
void detect_head_lvl(const float* __restrict__ x,
                     const float* __restrict__ bb, const float* __restrict__ bc,
                     const _Float16* __restrict__ wpk,
                     float* __restrict__ out)
{
    constexpr int nkc = C / 32;
    constexpr float lstride = (float)STRIDE_I;

    // Wave-private LDS scratch: box logits (64 rows x 16 anchors) + 4 decoded dists.
    __shared__ float lds_box[WPB][64][16];
    __shared__ float lds_dist[WPB][4][16];

    const int lane = threadIdx.x & 31;
    const int wv   = threadIdx.x >> 5;
    const int col  = lane & 15;
    const int hiK  = lane >> 4;

    const int T = blockIdx.x * WPB + wv;
    if (T >= NTILES) return;                // whole-wave uniform exit (EXEC stays all-ones)
    const int bat = blockIdx.y;
    const int a0  = T * 16;                 // anchor base, local to this level

    const float*    xbase = x + (size_t)bat * C * HW + (a0 + col);
    const _Float16* wbase = wpk + (size_t)lane * 32;

    v8f acc[9];
    #pragma unroll
    for (int m = 0; m < 9; ++m) { v8f z = {}; acc[m] = z; }

    for (int kc = 0; kc < nkc; ++kc) {
        // ---- B fragment (x tile, 32x16): lane holds K = kc*32 + hiK*8 + (e<8 ? e : e+8), col = anchor
        v16h Bh, Bl;
        const float* xp = xbase + (size_t)(kc * 32 + hiK * 8) * HW;
        #pragma unroll
        for (int e = 0; e < 16; ++e) {
            float v = xp[(size_t)(((e < 8) ? e : (e + 8)) * HW)];          // immediate offsets (HW constexpr)
            _Float16 h = (_Float16)v;
            Bh[e] = h;
            Bl[e] = (_Float16)(v - (float)h);
        }

        // ---- 9 M-tiles of pre-split weights: b128 loads w/ immediate offsets, no VALU
        const _Float16* ap0 = wbase + (size_t)kc * (32 * 32);
        #pragma unroll
        for (int m = 0; m < 9; ++m) {
            const _Float16* ap = ap0 + (size_t)(m * nkc * 32 * 32);  // constexpr offset
            v16h Ah = *(const v16h*)(ap);
            v16h Al = *(const v16h*)(ap + 16);
            // fp32-accuracy split product: hi*hi + hi*lo + lo*hi (f32 accumulate)
            acc[m] = __builtin_amdgcn_wmma_f32_16x16x32_f16(false, Ah, false, Bh, (short)0, acc[m], false, false);
            acc[m] = __builtin_amdgcn_wmma_f32_16x16x32_f16(false, Ah, false, Bl, (short)0, acc[m], false, false);
            acc[m] = __builtin_amdgcn_wmma_f32_16x16x32_f16(false, Al, false, Bh, (short)0, acc[m], false, false);
        }
    }

    // ---- bias add (C/D layout: row = i + 8*hiK, col = lane&15)
    #pragma unroll
    for (int m = 0; m < 9; ++m) {
        #pragma unroll
        for (int i = 0; i < 8; ++i) {
            int o = m * 16 + i + hiK * 8;
            acc[m][i] += (m < 4) ? bb[o] : bc[o - 64];
        }
    }

    // ---- DFL: box logits -> wave-private LDS (same-wave DS ordering, no barrier needed)
    #pragma unroll
    for (int m = 0; m < 4; ++m)
        #pragma unroll
        for (int i = 0; i < 8; ++i)
            lds_box[wv][m * 16 + i + hiK * 8][col] = acc[m][i];

    // each lane computes sides k = hiK and hiK+2 for its column: softmax-expectation, 16 bins
    #pragma unroll
    for (int t = 0; t < 2; ++t) {
        int k = hiK + 2 * t;
        float mx = lds_box[wv][k * 16][col];
        #pragma unroll
        for (int r = 1; r < 16; ++r) mx = fmaxf(mx, lds_box[wv][k * 16 + r][col]);
        float s = 0.f, d = 0.f;
        #pragma unroll
        for (int r = 0; r < 16; ++r) {
            float e = __expf(lds_box[wv][k * 16 + r][col] - mx);
            s += e; d += (float)r * e;
        }
        lds_dist[wv][k][col] = d / s;
    }

    // ---- anchor decode + box store (one anchor per lane 0..15)
    if (lane < 16) {
        float l  = lds_dist[wv][0][lane];
        float tp = lds_dist[wv][1][lane];
        float r  = lds_dist[wv][2][lane];
        float bm = lds_dist[wv][3][lane];
        int al = a0 + lane;                       // local anchor
        float ax = (float)(al % W) + 0.5f;        // W constexpr -> strength-reduced
        float ay = (float)(al / W) + 0.5f;
        v4f box = { (ax + 0.5f * (r - l))   * lstride,
                    (ay + 0.5f * (bm - tp)) * lstride,
                    (l + r)   * lstride,
                    (tp + bm) * lstride };
        *(v4f*)(out + ((size_t)bat * A_TOTAL + (LSTART + al)) * 84) = box;
    }

    // ---- class scores straight from registers: j = (m-4)*16 + i + 8*hiK, contiguous per lane
    {
        float* op = out + ((size_t)bat * A_TOTAL + (LSTART + a0 + col)) * 84 + 4;
        #pragma unroll
        for (int m = 4; m < 9; ++m) {
            int j0 = (m - 4) * 16 + hiK * 8;
            v4f c0 = { acc[m][0], acc[m][1], acc[m][2], acc[m][3] };
            v4f c1 = { acc[m][4], acc[m][5], acc[m][6], acc[m][7] };
            *(v4f*)(op + j0)     = c0;
            *(v4f*)(op + j0 + 4) = c1;
        }
    }
}

extern "C" void kernel_launch(void* const* d_in, const int* in_sizes, int n_in,
                              void* d_out, int out_size, void* d_ws, size_t ws_size,
                              hipStream_t stream) {
    (void)in_sizes; (void)n_in; (void)out_size; (void)ws_size;

    _Float16* wp = (_Float16*)d_ws;
    float*    out = (float*)d_out;

    // Pre-pass: pack weights into A-fragment-ordered f16 hi/lo (720 KB total in d_ws).
    pack_weights<<<dim3((2304 + 255) / 256), dim3(256), 0, stream>>>(
        (const float*)d_in[3],  (const float*)d_in[5],  wp + WP0_OFF, 256);
    pack_weights<<<dim3((4608 + 255) / 256), dim3(256), 0, stream>>>(
        (const float*)d_in[7],  (const float*)d_in[9],  wp + WP1_OFF, 512);
    pack_weights<<<dim3((4608 + 255) / 256), dim3(256), 0, stream>>>(
        (const float*)d_in[11], (const float*)d_in[13], wp + WP2_OFF, 512);

    // Level 0: 80x80, C=256, stride 8 -> 400 tiles, 50 blocks
    detect_head_lvl<256, 6400, 80, 8, 0, 400><<<dim3(50, 16), dim3(256), 0, stream>>>(
        (const float*)d_in[0], (const float*)d_in[4], (const float*)d_in[6],
        wp + WP0_OFF, out);
    // Level 1: 40x40, C=512, stride 16 -> 100 tiles, 13 blocks (last block: 4 idle waves)
    detect_head_lvl<512, 1600, 40, 16, 6400, 100><<<dim3(13, 16), dim3(256), 0, stream>>>(
        (const float*)d_in[1], (const float*)d_in[8], (const float*)d_in[10],
        wp + WP1_OFF, out);
    // Level 2: 20x20, C=512, stride 32 -> 25 tiles, 4 blocks (last block: 7 idle waves)
    detect_head_lvl<512, 400, 20, 32, 8000, 25><<<dim3(4, 16), dim3(256), 0, stream>>>(
        (const float*)d_in[2], (const float*)d_in[12], (const float*)d_in[14],
        wp + WP2_OFF, out);
}
